// Gemma4TextDecoderLayer_30288109371942
// MI455X (gfx1250) — compile-verified, matrix operand order checked by reading, not run on hardware
//
#include <hip/hip_runtime.h>
#include <hip/hip_bf16.h>
#include <math.h>

// ---- problem constants (from reference) ----
#define NHH   16
#define NKVH  8
#define HDSZ  256
#define SLEN  512
#define HDIM  2816
#define IDIM  704
#define NEXP  32
#define TOPK  4
#define QDIM  (NHH * HDSZ)   // 4096
#define KVDIM (NKVH * HDSZ)  // 2048
#define MAXTILE 96           // sum ceil(cnt_e/32) <= 2048/32 + 31 = 95
#define TOTSLOT 3072         // 2048 + 31*32 = 3040, rounded

typedef __attribute__((ext_vector_type(16))) __bf16 v16bf;
typedef __attribute__((ext_vector_type(8)))  float  v8f;

// f32 memory -> bf16 fragment (weight side: stream + in-register convert)
__device__ __forceinline__ v16bf load_frag_f32(const float* __restrict__ p) {
  v16bf r;
#pragma unroll
  for (int i = 0; i < 16; ++i) r[i] = (__bf16)p[i];
  return r;
}

// bf16 memory -> fragment (32B aligned)
__device__ __forceinline__ v16bf load_frag_bf(const __bf16* __restrict__ p) {
  return *(const v16bf*)p;
}

__device__ __forceinline__ v8f wmma_bf16(v16bf a, v16bf b, v8f c) {
  return __builtin_amdgcn_wmma_f32_16x16x32_bf16(false, a, false, b, (short)0, c,
                                                 false, false);
}

__device__ __forceinline__ float gelu_tanh(float x) {
  float x3 = x * x * x;
  return 0.5f * x * (1.0f + tanhf(0.79788456080286535588f * (x + 0.044715f * x3)));
}

// ---------------------------------------------------------------------------
// out[row] = (res ? res[row] : 0) + rmsnorm(in[row]) * w  (+ optional bf16 mirror)
// ---------------------------------------------------------------------------
__global__ void __launch_bounds__(256) add_rmsnorm_kernel(
    const float* __restrict__ res, const float* __restrict__ in,
    const float* __restrict__ w, float* __restrict__ out,
    __bf16* __restrict__ outb, int Hn) {
  const int row = blockIdx.x, tid = threadIdx.x;
  const float* xr = in + (size_t)row * Hn;
  __shared__ float red[8];
  float ss = 0.0f;
  for (int i = tid; i < Hn; i += 256) { float v = xr[i]; ss += v * v; }
#pragma unroll
  for (int m = 16; m >= 1; m >>= 1) ss += __shfl_xor(ss, m, 32);
  if ((tid & 31) == 0) red[tid >> 5] = ss;
  __syncthreads();
  float tot = 0.0f;
#pragma unroll
  for (int j = 0; j < 8; ++j) tot += red[j];
  const float rs = rsqrtf(tot / (float)Hn + 1e-6f);
  for (int i = tid; i < Hn; i += 256) {
    float v = xr[i] * rs * w[i];
    float base = res ? res[(size_t)row * Hn + i] : 0.0f;
    float o = base + v;
    if (out)  out[(size_t)row * Hn + i] = o;
    if (outb) outb[(size_t)row * Hn + i] = (__bf16)o;
  }
}

// ---------------------------------------------------------------------------
// Dense GEMM: C[M,N] = A[M,K](bf16) @ B[N,K](f32)^T.
// block = 128 (4 waves); block tile 128x64. The 64x32 B tile is staged
// through LDS once per k-step (f32 loaded + converted once per block instead
// of once per wave), each wave then reads bf16 fragments via ds_load.
// ---------------------------------------------------------------------------
__global__ void __launch_bounds__(128) gemm_bf16_nt(
    const __bf16* __restrict__ A, const float* __restrict__ B,
    float* __restrict__ C, int N, int K) {
  __shared__ __align__(32) __bf16 btile[64 * 32];  // [col][k], 4KB
  const int tid = threadIdx.x;
  const int lane = tid & 31, wv = tid >> 5;
  const int row0 = blockIdx.y * 128 + wv * 32;
  const int col0 = blockIdx.x * 64;
  const int r = lane & 15, hi = lane >> 4;
  // staging role: col = tid>>1 (0..63), k-half = tid&1
  const int scol = tid >> 1, skh = tid & 1;
  const float* bsrc = B + (size_t)(col0 + scol) * K + skh * 16;
  __bf16* bdst = btile + scol * 32 + skh * 16;

  v8f acc[2][4];
#pragma unroll
  for (int t = 0; t < 2; ++t)
#pragma unroll
    for (int s = 0; s < 4; ++s) acc[t][s] = (v8f){};
  const __bf16* a0 = A + (size_t)(row0 + r) * K + hi * 16;
  const __bf16* a1 = a0 + (size_t)16 * K;

  for (int k = 0; k < K; k += 32) {
    *(v16bf*)bdst = load_frag_f32(bsrc + k);   // stage + convert once
    __syncthreads();
    v16bf af0 = load_frag_bf(a0 + k);
    v16bf af1 = load_frag_bf(a1 + k);
#pragma unroll
    for (int s = 0; s < 4; ++s) {
      v16bf bf = *(const v16bf*)(btile + (s * 16 + r) * 32 + hi * 16);
      acc[0][s] = wmma_bf16(af0, bf, acc[0][s]);
      acc[1][s] = wmma_bf16(af1, bf, acc[1][s]);
    }
    __syncthreads();
  }
#pragma unroll
  for (int t = 0; t < 2; ++t)
#pragma unroll
    for (int i = 0; i < 8; ++i) {
      float* cr = C + (size_t)(row0 + t * 16 + i + 8 * hi) * N + col0 + r;
      cr[0] = acc[t][0][i]; cr[16] = acc[t][1][i];
      cr[32] = acc[t][2][i]; cr[48] = acc[t][3][i];
    }
}

// ---------------------------------------------------------------------------
// Per-head RMSNorm + RoPE -> bf16 q, k, transposed v.
// grid = (SLEN, NHH), block = 32.
// ---------------------------------------------------------------------------
__global__ void __launch_bounds__(32) qkv_post_kernel(
    const float* __restrict__ qin, const float* __restrict__ kin,
    const float* __restrict__ vin, __bf16* __restrict__ qbf,
    __bf16* __restrict__ kbf, __bf16* __restrict__ vTb,
    const float* __restrict__ cosb, const float* __restrict__ sinb,
    const float* __restrict__ qnw, const float* __restrict__ knw) {
  const int t = blockIdx.x, h = blockIdx.y, lane = threadIdx.x;
  const float* ct = cosb + (size_t)t * HDSZ;
  const float* st = sinb + (size_t)t * HDSZ;
  {
    const float* p = qin + (size_t)t * QDIM + h * HDSZ;
    float x[8]; float ss = 0.0f;
#pragma unroll
    for (int i = 0; i < 8; ++i) { x[i] = p[lane * 8 + i]; ss += x[i] * x[i]; }
#pragma unroll
    for (int m = 16; m >= 1; m >>= 1) ss += __shfl_xor(ss, m, 32);
    const float rs = rsqrtf(ss * (1.0f / HDSZ) + 1e-6f);
    float y[8];
#pragma unroll
    for (int i = 0; i < 8; ++i) y[i] = x[i] * rs * qnw[lane * 8 + i];
#pragma unroll
    for (int i = 0; i < 8; ++i) {
      const float yp = __shfl_xor(y[i], 16, 32);
      const int d = lane * 8 + i;
      const float c = ct[d], s = st[d];
      const float v = (lane < 16) ? (y[i] * c - yp * s) : (y[i] * c + yp * s);
      qbf[(size_t)t * QDIM + h * HDSZ + d] = (__bf16)v;
    }
  }
  if (h < NKVH) {
    {
      const float* p = kin + (size_t)t * KVDIM + h * HDSZ;
      float x[8]; float ss = 0.0f;
#pragma unroll
      for (int i = 0; i < 8; ++i) { x[i] = p[lane * 8 + i]; ss += x[i] * x[i]; }
#pragma unroll
      for (int m = 16; m >= 1; m >>= 1) ss += __shfl_xor(ss, m, 32);
      const float rs = rsqrtf(ss * (1.0f / HDSZ) + 1e-6f);
      float y[8];
#pragma unroll
      for (int i = 0; i < 8; ++i) y[i] = x[i] * rs * knw[lane * 8 + i];
#pragma unroll
      for (int i = 0; i < 8; ++i) {
        const float yp = __shfl_xor(y[i], 16, 32);
        const int d = lane * 8 + i;
        const float c = ct[d], s = st[d];
        const float v = (lane < 16) ? (y[i] * c - yp * s) : (y[i] * c + yp * s);
        kbf[(size_t)t * KVDIM + h * HDSZ + d] = (__bf16)v;
      }
    }
    {
      const float* p = vin + (size_t)t * KVDIM + h * HDSZ;
      float x[8]; float ss = 0.0f;
#pragma unroll
      for (int i = 0; i < 8; ++i) { x[i] = p[lane * 8 + i]; ss += x[i] * x[i]; }
#pragma unroll
      for (int m = 16; m >= 1; m >>= 1) ss += __shfl_xor(ss, m, 32);
      const float rs = rsqrtf(ss * (1.0f / HDSZ) + 1e-6f);
#pragma unroll
      for (int i = 0; i < 8; ++i) {
        const int d = lane * 8 + i;
        vTb[((size_t)h * HDSZ + d) * SLEN + t] = (__bf16)(x[i] * rs);
      }
    }
  }
}

// ---------------------------------------------------------------------------
// Causal flash attention; grid = (SLEN/16, NHH); block = 32 (one wave).
// ---------------------------------------------------------------------------
__global__ void __launch_bounds__(32) attn_kernel(
    const __bf16* __restrict__ q, const __bf16* __restrict__ k,
    const __bf16* __restrict__ vT, __bf16* __restrict__ o) {
  const int lane = threadIdx.x;
  const int qt = blockIdx.x, head = blockIdx.y, kvh = head >> 1;
  const int qbase = qt * 16;
  const int r = lane & 15, hi = lane >> 4;
  __shared__ float ptile[16 * 32];

  v16bf qf[8];
  const __bf16* qrow = q + (size_t)(qbase + r) * QDIM + head * HDSZ + hi * 16;
#pragma unroll
  for (int kd = 0; kd < 8; ++kd) qf[kd] = load_frag_bf(qrow + kd * 32);

  v8f oacc[16];
#pragma unroll
  for (int dt = 0; dt < 16; ++dt) oacc[dt] = (v8f){};
  float mrow[8], lrow[8];
#pragma unroll
  for (int i = 0; i < 8; ++i) { mrow[i] = -3.0e38f; lrow[i] = 0.0f; }

  const int nchunk = qbase / 32 + 1;
  for (int c = 0; c < nchunk; ++c) {
    const int jb = c * 32;
    v8f s0 = (v8f){}, s1 = (v8f){};
    const __bf16* k0 = k + (size_t)(jb + r) * KVDIM + kvh * HDSZ + hi * 16;
    const __bf16* k1 = k0 + (size_t)16 * KVDIM;
#pragma unroll
    for (int kd = 0; kd < 8; ++kd) {
      s0 = wmma_bf16(qf[kd], load_frag_bf(k0 + kd * 32), s0);
      s1 = wmma_bf16(qf[kd], load_frag_bf(k1 + kd * 32), s1);
    }
#pragma unroll
    for (int i = 0; i < 8; ++i) {
      const int row = qbase + i + 8 * hi;
      float v0 = (jb + r      <= row) ? s0[i] : -3.0e38f;
      float v1 = (jb + 16 + r <= row) ? s1[i] : -3.0e38f;
      float mx = fmaxf(v0, v1);
#pragma unroll
      for (int m = 8; m >= 1; m >>= 1) mx = fmaxf(mx, __shfl_xor(mx, m, 16));
      const float mnew  = fmaxf(mrow[i], mx);
      const float alpha = __expf(mrow[i] - mnew);
      const float p0 = __expf(v0 - mnew);
      const float p1 = __expf(v1 - mnew);
      float ps = p0 + p1;
#pragma unroll
      for (int m = 8; m >= 1; m >>= 1) ps += __shfl_xor(ps, m, 16);
      lrow[i] = lrow[i] * alpha + ps;
      mrow[i] = mnew;
#pragma unroll
      for (int dt = 0; dt < 16; ++dt) oacc[dt][i] *= alpha;
      ptile[(i + 8 * hi) * 32 + r]      = p0;
      ptile[(i + 8 * hi) * 32 + 16 + r] = p1;
    }
    __syncthreads();
    v16bf pf;
    {
      const float* pl = ptile + r * 32 + hi * 16;
#pragma unroll
      for (int i = 0; i < 16; ++i) pf[i] = (__bf16)pl[i];
    }
    __syncthreads();
    const __bf16* vbase = vT + (size_t)kvh * HDSZ * SLEN + jb + hi * 16;
#pragma unroll
    for (int dt = 0; dt < 16; ++dt) {
      const __bf16* vp = vbase + (size_t)(dt * 16 + r) * SLEN;
      oacc[dt] = wmma_bf16(pf, load_frag_bf(vp), oacc[dt]);
    }
  }
#pragma unroll
  for (int dt = 0; dt < 16; ++dt)
#pragma unroll
    for (int i = 0; i < 8; ++i)
      o[(size_t)(qbase + i + 8 * hi) * QDIM + head * HDSZ + dt * 16 + r] =
          (__bf16)(oacc[dt][i] / lrow[i]);
}

// ---------------------------------------------------------------------------
// Router softmax + top-4; grid = SLEN, block = 32 (lane == expert).
// ---------------------------------------------------------------------------
__global__ void __launch_bounds__(32) router_topk_kernel(
    const float* __restrict__ h2, const float* __restrict__ rw,
    const float* __restrict__ rscale, float* __restrict__ comb) {
  const int t = blockIdx.x, e = threadIdx.x;
  const float* x = h2 + (size_t)t * HDIM;
  float ss = 0.0f;
  for (int i = e; i < HDIM; i += 32) ss += x[i] * x[i];
#pragma unroll
  for (int m = 16; m >= 1; m >>= 1) ss += __shfl_xor(ss, m, 32);
  const float rs = rsqrtf(ss / (float)HDIM + 1e-6f);
  const float hscale = rsqrtf((float)HDIM);
  const float* w = rw + (size_t)e * HDIM;
  float dot = 0.0f;
  for (int i = 0; i < HDIM; ++i)
    dot += (x[i] * rs * hscale * rscale[i]) * w[i];
  float mx = dot;
#pragma unroll
  for (int m = 16; m >= 1; m >>= 1) mx = fmaxf(mx, __shfl_xor(mx, m, 32));
  float p = __expf(dot - mx);
  float sum = p;
#pragma unroll
  for (int m = 16; m >= 1; m >>= 1) sum += __shfl_xor(sum, m, 32);
  p /= sum;
  float pw = p;
  bool sel = false;
  float topsum = 0.0f;
  for (int it = 0; it < TOPK; ++it) {
    float v = pw; int idx = e;
#pragma unroll
    for (int m = 16; m >= 1; m >>= 1) {
      float ov = __shfl_xor(v, m, 32);
      int   oi = __shfl_xor(idx, m, 32);
      if (ov > v || (ov == v && oi < idx)) { v = ov; idx = oi; }
    }
    if (e == idx) { sel = true; pw = -1.0f; }
    topsum += v;
  }
  comb[(size_t)t * NEXP + e] = sel ? (p / topsum) : 0.0f;
}

// ---------------------------------------------------------------------------
// Token gather: per-expert ordered compaction + padded tile table.
// ONE block of 1024 threads (wave e handles expert e). Deterministic.
// ---------------------------------------------------------------------------
__global__ void __launch_bounds__(1024) moe_gather_kernel(
    const float* __restrict__ comb, int* __restrict__ cnt,
    int* __restrict__ off, int* __restrict__ tok,
    int* __restrict__ tileE, int* __restrict__ tileR) {
  const int e = threadIdx.x >> 5;
  const int lane = threadIdx.x & 31;
  __shared__ int scnt[NEXP], soff[NEXP];
  int c = 0;
  for (int t = lane; t < SLEN; t += 32) c += (comb[(size_t)t * NEXP + e] > 0.0f);
#pragma unroll
  for (int m = 16; m >= 1; m >>= 1) c += __shfl_xor(c, m, 32);
  if (lane == 0) scnt[e] = c;
  __syncthreads();
  if (threadIdx.x == 0) {
    int o = 0, tt = 0;
    for (int i = 0; i < NEXP; ++i) {
      soff[i] = o;
      const int nt = (scnt[i] + 31) / 32;
      for (int j = 0; j < nt; ++j) { tileE[tt] = i; tileR[tt] = o + j * 32; ++tt; }
      o += nt * 32;
    }
    for (int j = tt; j < MAXTILE; ++j) { tileE[j] = -1; tileR[j] = 0; }
  }
  __syncthreads();
  int base = soff[e];
  for (int t0 = 0; t0 < SLEN; t0 += 32) {
    const int t = t0 + lane;
    const bool f = comb[(size_t)t * NEXP + e] > 0.0f;
    const unsigned long long mask = __ballot(f);
    const int pos = __popcll(mask & ((1ull << lane) - 1ull));
    if (f) tok[base + pos] = t;
    base += __popcll(mask);
  }
  if (lane == 0) { cnt[e] = scnt[e]; off[e] = soff[e]; }
}

// ---------------------------------------------------------------------------
// Gathered gate/up: acmp[slot] = gelu(h2b[tok]@Wg^T) * (h2b[tok]@Wu^T), bf16.
// grid = (ceil(IDIM/256)=3, MAXTILE); block = 128; each wave = same 32-slot
// tile, its own 64-column group (B fragments wave-private: no redundancy).
// ---------------------------------------------------------------------------
__global__ void __launch_bounds__(128) moe_gateup_kernel(
    const __bf16* __restrict__ A, const float* __restrict__ Wg,
    const float* __restrict__ Wu, const int* __restrict__ cnt,
    const int* __restrict__ off, const int* __restrict__ tok,
    const int* __restrict__ tileE, const int* __restrict__ tileR,
    __bf16* __restrict__ acmp) {
  const int e = tileE[blockIdx.y];
  if (e < 0) return;
  const int lane = threadIdx.x & 31, wv = threadIdx.x >> 5;
  const int col0 = blockIdx.x * 256 + wv * 64;
  if (col0 >= IDIM) return;          // wave-level exit; no barriers used
  const int slot0 = tileR[blockIdx.y];
  const int r = lane & 15, hi = lane >> 4;
  const int offe = off[e], hiend = offe + cnt[e];
  const int sA = slot0 + r, sB = slot0 + 16 + r;
  const int tok0 = (sA < hiend) ? tok[sA] : tok[offe];
  const int tok1 = (sB < hiend) ? tok[sB] : tok[offe];
  const __bf16* a0 = A + (size_t)tok0 * HDIM + hi * 16;
  const __bf16* a1 = A + (size_t)tok1 * HDIM + hi * 16;
  const float* wg = Wg + (size_t)e * IDIM * HDIM;
  const float* wu = Wu + (size_t)e * IDIM * HDIM;
  v8f g[2][4], u[2][4];
#pragma unroll
  for (int t = 0; t < 2; ++t)
#pragma unroll
    for (int s = 0; s < 4; ++s) { g[t][s] = (v8f){}; u[t][s] = (v8f){}; }
  for (int kk = 0; kk < HDIM; kk += 32) {
    v16bf af0 = load_frag_bf(a0 + kk);
    v16bf af1 = load_frag_bf(a1 + kk);
#pragma unroll
    for (int s = 0; s < 4; ++s) {
      const size_t boff = (size_t)(col0 + s * 16 + r) * HDIM + hi * 16 + kk;
      v16bf bg = load_frag_f32(wg + boff);
      g[0][s] = wmma_bf16(af0, bg, g[0][s]);
      g[1][s] = wmma_bf16(af1, bg, g[1][s]);
      v16bf bu = load_frag_f32(wu + boff);
      u[0][s] = wmma_bf16(af0, bu, u[0][s]);
      u[1][s] = wmma_bf16(af1, bu, u[1][s]);
    }
  }
#pragma unroll
  for (int t = 0; t < 2; ++t)
#pragma unroll
    for (int s = 0; s < 4; ++s)
#pragma unroll
      for (int i = 0; i < 8; ++i) {
        const float val = gelu_tanh(g[t][s][i]) * u[t][s][i];
        acmp[(size_t)(slot0 + t * 16 + i + 8 * hi) * IDIM + col0 + s * 16 + r] =
            (__bf16)val;
      }
}

// ---------------------------------------------------------------------------
// Gathered down-proj: y[slot] = acmp[slot] @ Wd[e]^T (f32 rows, per slot).
// grid = (HDIM/256=11, MAXTILE); block = 128; wave = 64-column group.
// ---------------------------------------------------------------------------
__global__ void __launch_bounds__(128) moe_down_kernel(
    const __bf16* __restrict__ acmp, const float* __restrict__ Wd,
    const int* __restrict__ tileE, const int* __restrict__ tileR,
    float* __restrict__ y) {
  const int e = tileE[blockIdx.y];
  if (e < 0) return;
  const int lane = threadIdx.x & 31, wv = threadIdx.x >> 5;
  const int col0 = blockIdx.x * 256 + wv * 64;
  const int slot0 = tileR[blockIdx.y];
  const int r = lane & 15, hi = lane >> 4;
  const __bf16* a0 = acmp + (size_t)(slot0 + r) * IDIM + hi * 16;
  const __bf16* a1 = a0 + (size_t)16 * IDIM;
  const float* b0 = Wd + (size_t)e * HDIM * IDIM +
                    (size_t)(col0 + r) * IDIM + hi * 16;
  v8f acc[2][4];
#pragma unroll
  for (int t = 0; t < 2; ++t)
#pragma unroll
    for (int s = 0; s < 4; ++s) acc[t][s] = (v8f){};
  for (int kk = 0; kk < IDIM; kk += 32) {
    v16bf af0 = load_frag_bf(a0 + kk);
    v16bf af1 = load_frag_bf(a1 + kk);
#pragma unroll
    for (int s = 0; s < 4; ++s) {
      v16bf bf = load_frag_f32(b0 + (size_t)(s * 16) * IDIM + kk);
      acc[0][s] = wmma_bf16(af0, bf, acc[0][s]);
      acc[1][s] = wmma_bf16(af1, bf, acc[1][s]);
    }
  }
#pragma unroll
  for (int t = 0; t < 2; ++t)
#pragma unroll
    for (int i = 0; i < 8; ++i) {
      float* cr = y + (size_t)(slot0 + t * 16 + i + 8 * hi) * HDIM + col0 + r;
      cr[0] = acc[t][0][i]; cr[16] = acc[t][1][i];
      cr[32] = acc[t][2][i]; cr[48] = acc[t][3][i];
    }
}

// ---------------------------------------------------------------------------
// Combine: mo[t] = sum of comb[t,e] * y[slot(t,e)] over the 4 selected
// experts; slot found by binary search in the sorted per-expert token list.
// grid = SLEN, block = 256. Fully deterministic (no atomics).
// ---------------------------------------------------------------------------
__global__ void __launch_bounds__(256) moe_combine_kernel(
    const float* __restrict__ comb, const int* __restrict__ cnt,
    const int* __restrict__ off, const int* __restrict__ tok,
    const float* __restrict__ y, float* __restrict__ mo) {
  const int t = blockIdx.x, tid = threadIdx.x;
  int   ss_[TOPK];
  float sw_[TOPK];
  int ns = 0;
  for (int e = 0; e < NEXP; ++e) {
    const float w = comb[(size_t)t * NEXP + e];
    if (w > 0.0f && ns < TOPK) {
      int lo = off[e], hi2 = off[e] + cnt[e];
      while (lo < hi2) {
        const int mid = (lo + hi2) >> 1;
        if (tok[mid] < t) lo = mid + 1; else hi2 = mid;
      }
      ss_[ns] = lo; sw_[ns] = w; ++ns;
    }
  }
  for (int i = tid; i < HDIM; i += 256) {
    float acc = 0.0f;
    for (int j = 0; j < ns; ++j) acc += sw_[j] * y[(size_t)ss_[j] * HDIM + i];
    mo[(size_t)t * HDIM + i] = acc;
  }
}

// ---------------------------------------------------------------------------
extern "C" void kernel_launch(void* const* d_in, const int* in_sizes, int n_in,
                              void* d_out, int out_size, void* d_ws,
                              size_t ws_size, hipStream_t stream) {
  (void)in_sizes; (void)n_in; (void)out_size; (void)ws_size;
  const float* hidden         = (const float*)d_in[0];
  const float* cosb           = (const float*)d_in[1];
  const float* sinb           = (const float*)d_in[2];
  const float* ln_in_w        = (const float*)d_in[3];
  const float* ln_post_attn_w = (const float*)d_in[4];
  const float* ln_pre_ffn_w   = (const float*)d_in[5];
  const float* ln_post_ffn_w  = (const float*)d_in[6];
  const float* q_w            = (const float*)d_in[7];
  const float* k_w            = (const float*)d_in[8];
  const float* v_w            = (const float*)d_in[9];
  const float* o_w            = (const float*)d_in[10];
  const float* q_norm_w       = (const float*)d_in[11];
  const float* k_norm_w       = (const float*)d_in[12];
  const float* router_w       = (const float*)d_in[13];
  const float* router_scale   = (const float*)d_in[14];
  const float* w_gate         = (const float*)d_in[15];
  const float* w_up           = (const float*)d_in[16];
  const float* w_down         = (const float*)d_in[17];
  float* out = (float*)d_out;

  char* base = (char*)d_ws;
  size_t off = 0;
  auto alloc = [&](size_t bytes) -> void* {
    void* p = base + off;
    off += (bytes + 255) & ~(size_t)255;
    return p;
  };
  __bf16* hnb = (__bf16*)alloc((size_t)SLEN * HDIM * 2);
  float*  qb  = (float*)alloc((size_t)SLEN * QDIM * 4);
  float*  kb  = (float*)alloc((size_t)SLEN * KVDIM * 4);
  float*  vb  = (float*)alloc((size_t)SLEN * KVDIM * 4);
  __bf16* qbf = (__bf16*)alloc((size_t)SLEN * QDIM * 2);
  __bf16* kbf = (__bf16*)alloc((size_t)SLEN * KVDIM * 2);
  __bf16* vTb = (__bf16*)alloc((size_t)NKVH * HDSZ * SLEN * 2);
  __bf16* aob = (__bf16*)alloc((size_t)SLEN * QDIM * 2);
  float*  ot  = (float*)alloc((size_t)SLEN * HDIM * 4);
  float*  xb  = (float*)alloc((size_t)SLEN * HDIM * 4);
  float*  h2  = (float*)alloc((size_t)SLEN * HDIM * 4);
  __bf16* h2b = (__bf16*)alloc((size_t)SLEN * HDIM * 2);
  float*  cb  = (float*)alloc((size_t)SLEN * NEXP * 4);
  int*    icnt = (int*)alloc(NEXP * 4);
  int*    ioff = (int*)alloc(NEXP * 4);
  int*    itok = (int*)alloc(TOTSLOT * 4);
  int*    itE  = (int*)alloc(MAXTILE * 4);
  int*    itR  = (int*)alloc(MAXTILE * 4);
  __bf16* acmp = (__bf16*)alloc((size_t)TOTSLOT * IDIM * 2);
  float*  ybuf = (float*)alloc((size_t)TOTSLOT * HDIM * 4);
  float*  mo   = (float*)alloc((size_t)SLEN * HDIM * 4);

  // 1. h = rmsnorm(hidden, ln_in_w) -> bf16
  add_rmsnorm_kernel<<<dim3(SLEN), dim3(256), 0, stream>>>(
      (const float*)nullptr, hidden, ln_in_w, (float*)nullptr, hnb, HDIM);
  // 2. Q/K/V projections (LDS-staged bf16 WMMA)
  gemm_bf16_nt<<<dim3(QDIM / 64, SLEN / 128), dim3(128), 0, stream>>>(
      hnb, q_w, qb, QDIM, HDIM);
  gemm_bf16_nt<<<dim3(KVDIM / 64, SLEN / 128), dim3(128), 0, stream>>>(
      hnb, k_w, kb, KVDIM, HDIM);
  gemm_bf16_nt<<<dim3(KVDIM / 64, SLEN / 128), dim3(128), 0, stream>>>(
      hnb, v_w, vb, KVDIM, HDIM);
  // 3. head rmsnorm + RoPE -> bf16 q/k, transposed bf16 v
  qkv_post_kernel<<<dim3(SLEN, NHH), dim3(32), 0, stream>>>(
      qb, kb, vb, qbf, kbf, vTb, cosb, sinb, q_norm_w, k_norm_w);
  // 4. causal flash attention
  attn_kernel<<<dim3(SLEN / 16, NHH), dim3(32), 0, stream>>>(
      qbf, kbf, vTb, aob);
  // 5. output projection
  gemm_bf16_nt<<<dim3(HDIM / 64, SLEN / 128), dim3(128), 0, stream>>>(
      aob, o_w, ot, HDIM, QDIM);
  // 6. x = hidden + rmsnorm(o, ln_post_attn_w)
  add_rmsnorm_kernel<<<dim3(SLEN), dim3(256), 0, stream>>>(
      hidden, ot, ln_post_attn_w, xb, (__bf16*)nullptr, HDIM);
  // 7. h2 = rmsnorm(x, ln_pre_ffn_w)
  add_rmsnorm_kernel<<<dim3(SLEN), dim3(256), 0, stream>>>(
      (const float*)nullptr, xb, ln_pre_ffn_w, h2, h2b, HDIM);
  // 8. router softmax + top-4
  router_topk_kernel<<<dim3(SLEN), dim3(32), 0, stream>>>(
      h2, router_w, router_scale, cb);
  // 9. deterministic token gather + tile table
  moe_gather_kernel<<<dim3(1), dim3(1024), 0, stream>>>(
      cb, icnt, ioff, itok, itE, itR);
  // 10. gathered gate/up (only ~2048 of 16384 dense rows)
  moe_gateup_kernel<<<dim3((IDIM + 255) / 256, MAXTILE), dim3(128), 0,
                     stream>>>(h2b, w_gate, w_up, icnt, ioff, itok, itE, itR,
                               acmp);
  // 11. gathered down-proj
  moe_down_kernel<<<dim3(HDIM / 256, MAXTILE), dim3(128), 0, stream>>>(
      acmp, w_down, itE, itR, ybuf);
  // 12. per-token combine (binary search, no atomics)
  moe_combine_kernel<<<dim3(SLEN), dim3(256), 0, stream>>>(
      cb, icnt, ioff, itok, ybuf, mo);
  // 13. out = x + rmsnorm(moe, ln_post_ffn_w)
  add_rmsnorm_kernel<<<dim3(SLEN), dim3(256), 0, stream>>>(
      xb, mo, ln_post_ffn_w, out, (__bf16*)nullptr, HDIM);
}
